// LayerWiseBaseSlidingPool_13700945674784
// MI455X (gfx1250) — matrix operand order repaired and checked by reading, not compiled, not run on hardware
//
#include <hip/hip_runtime.h>
#include <hip/hip_bf16.h>

#define B_ 16
#define L_ 13
#define T_ 1500
#define D_ 512
#define NMASK 450          // int(1500 * 0.3)
#define MTILE 64           // rows of X per workgroup
#define WKROWS 32          // K-slice of W staged per iteration
#define XPAD 8             // Xs row = 520 bf16 = 1040B (16B mult, bank-spread)
#define WPAD 8             // Ws row = 40 bf16 = 80B  (16B mult, bank-spread)
#define NCHUNKS 10
#define TCHUNK 150

typedef __attribute__((ext_vector_type(16))) __bf16 v16bf;
typedef __attribute__((ext_vector_type(8)))  float  v8f;
typedef __attribute__((ext_vector_type(4)))  unsigned int u32x4;

union Frag16 { v16bf bf; u32x4 q[2]; };

__device__ __forceinline__ unsigned int f2bf(float f) {
    unsigned int u = __float_as_uint(f);
    unsigned int r = u + 0x7FFFu + ((u >> 16) & 1u);   // round-to-nearest-even
    return r >> 16;
}
__device__ __forceinline__ unsigned int pack2bf(float lo, float hi) {
    return f2bf(lo) | (f2bf(hi) << 16);
}

// ---------------------------------------------------------------------------
// Kernel T: one-time lin_w fp32 [L,D,E] -> bf16 transposed Wt [L,E,D].
// Tiled 32x32 LDS transpose, coalesced both directions. ~20MB traffic total.
// ---------------------------------------------------------------------------
__global__ void __launch_bounds__(256)
wt_transpose_kernel(const float* __restrict__ Wlin,      // [L, D, E]
                    unsigned short* __restrict__ Wt)     // [L, E, D] bf16
{
    __shared__ float tile[32][33];
    const int l  = blockIdx.z;
    const int d0 = blockIdx.x * 32;
    const int e0 = blockIdx.y * 32;
    const int c  = threadIdx.x & 31;
    const int r0 = (threadIdx.x >> 5) * 4;

    #pragma unroll
    for (int j = 0; j < 4; ++j) {
        int r = r0 + j;
        tile[r][c] = Wlin[((size_t)l * D_ + d0 + r) * D_ + e0 + c];
    }
    __syncthreads();
    #pragma unroll
    for (int j = 0; j < 4; ++j) {
        int r = r0 + j;
        Wt[((size_t)l * D_ + e0 + r) * D_ + d0 + c] = (unsigned short)f2bf(tile[c][r]);
    }
}

// ---------------------------------------------------------------------------
// Kernel A: per-(b,l) scores via bf16 WMMA GEMM + fused tanh/attn_w epilogue.
//   scores[bl][t] = sum_e tanh((X[t,:]·W[l])[e] + lin_b[l,e]) * attn_w[l,e] + attn_b[l]
// W is consumed pre-converted/pre-transposed (Wt); its K-slice staging is a
// pure b128 copy, software-pipelined (prefetch regs) to overlap the WMMAs.
// ---------------------------------------------------------------------------
__global__ void __launch_bounds__(256)
score_gemm_kernel(const float* __restrict__ X,             // [B*L, T, D]
                  const unsigned short* __restrict__ Wt,   // [L, E, D] bf16
                  const float* __restrict__ lin_b,         // [L, D]
                  const float* __restrict__ attn_w,        // [L, D]
                  const float* __restrict__ attn_b,        // [L]
                  float* __restrict__ scores)              // [B*L, T]
{
    __shared__ __attribute__((aligned(16))) unsigned short Xs[MTILE][D_ + XPAD];
    __shared__ __attribute__((aligned(16))) unsigned short Ws[D_][WKROWS + WPAD];
    __shared__ float score_s[MTILE];

    const int tid  = threadIdx.x;
    const int lane = tid & 31;
    const int wave = tid >> 5;
    const int bl   = blockIdx.y;
    const int l    = bl % L_;
    const int t0   = blockIdx.x * MTILE;

    const int mtile = wave & 3;    // 16-row slab within the 64-row tile
    const int nhalf = wave >> 2;   // which 256-column half of E

    if (tid < MTILE) score_s[tid] = 0.0f;

    // ---- stage X tile: 64x512 fp32 -> bf16 LDS (packed, ds_store_b64) ----
    {
        const float4* Xg = (const float4*)(X + (size_t)bl * T_ * D_);
        #pragma unroll
        for (int i = 0; i < (MTILE * D_ / 4) / 256; ++i) {
            int flat = i * 256 + tid;
            int r  = flat >> 7;          // 128 float4 per row
            int c4 = flat & 127;
            int t  = t0 + r;
            float4 v = make_float4(0.f, 0.f, 0.f, 0.f);
            if (t < T_) v = Xg[(size_t)t * (D_ / 4) + c4];
            uint2 pk;
            pk.x = pack2bf(v.x, v.y);
            pk.y = pack2bf(v.z, v.w);
            *(uint2*)&Xs[r][c4 * 4] = pk;
        }
    }

    v8f acc[16];
    #pragma unroll
    for (int n = 0; n < 16; ++n) acc[n] = (v8f)0.0f;

    const int mrow = lane & 15;          // A: row within tile / B: column within chunk
    const int koff = (lane >> 4) * 8;    // A frag K sub-offset per ISA layout
    const int khlf = (lane >> 4) * 16;   // unused; B frag uses koff like A

    // W slice staging: each thread copies 8 x 16B; thread -> (e-row, 16B chunk)
    const unsigned short* WtL = Wt + (size_t)l * D_ * D_;
    const int se = tid >> 2;             // base e-row (stride 64 over i)
    const int sc = (tid & 3) * 8;        // 16B chunk within the 32-K slice

    u32x4 pre[8];
    #pragma unroll
    for (int i = 0; i < 8; ++i)
        pre[i] = *(const u32x4*)(WtL + (size_t)(se + i * 64) * D_ + sc);

    for (int kk = 0; kk < D_; kk += WKROWS) {
        __syncthreads();                 // X staged / previous Ws fully consumed
        #pragma unroll
        for (int i = 0; i < 8; ++i)
            *(u32x4*)&Ws[se + i * 64][sc] = pre[i];
        __syncthreads();

        if (kk + WKROWS < D_) {          // prefetch next K-slice (overlaps WMMAs)
            #pragma unroll
            for (int i = 0; i < 8; ++i)
                pre[i] = *(const u32x4*)(WtL + (size_t)(se + i * 64) * D_ + (kk + WKROWS) + sc);
        }

        // A fragment: 16x32 bf16, lanes 0-15 hold K 0-7|16-23, lanes 16-31 hold 8-15|24-31
        Frag16 A;
        const unsigned short* ap = &Xs[mtile * 16 + mrow][kk + koff];
        A.q[0] = *(const u32x4*)ap;
        A.q[1] = *(const u32x4*)(ap + 16);

        #pragma unroll
        for (int nc = 0; nc < 16; ++nc) {
            // B fragment: 32x16 bf16, lane = column, K 0-7|16-23 (lanes 0-15), 8-15|24-31 (lanes 16-31)
            Frag16 Bf;
            const unsigned short* bp = &Ws[nhalf * 256 + nc * 16 + mrow][koff];
            Bf.q[0] = *(const u32x4*)bp;
            Bf.q[1] = *(const u32x4*)(bp + 16);
            acc[nc] = __builtin_amdgcn_wmma_f32_16x16x32_bf16(
                false, A.bf, false, Bf.bf, (short)0, acc[nc], false, false);
        }
    }

    // ---- fused epilogue: tanh(acc + b) . attn_w, reduced over E ----
    float rowsum[8];
    #pragma unroll
    for (int j = 0; j < 8; ++j) rowsum[j] = 0.0f;

    const float* lb = lin_b + (size_t)l * D_;
    const float* aw = attn_w + (size_t)l * D_;

    #pragma unroll
    for (int nc = 0; nc < 16; ++nc) {
        int e = nhalf * 256 + nc * 16 + mrow;
        float bias = lb[e];
        float wgt  = aw[e];
        #pragma unroll
        for (int j = 0; j < 8; ++j) {
            // C layout: vgpr j = row (j or j+8), lane%16 = column
            float v = tanhf(acc[nc][j] + bias) * wgt;
            v += __shfl_xor(v, 1, 16);
            v += __shfl_xor(v, 2, 16);
            v += __shfl_xor(v, 4, 16);
            v += __shfl_xor(v, 8, 16);
            rowsum[j] += v;            // butterfly: every lane holds group sum
        }
    }
    if ((lane & 15) == 0) {
        int mbase = mtile * 16 + ((lane >> 4) ? 8 : 0);
        #pragma unroll
        for (int j = 0; j < 8; ++j)
            atomicAdd(&score_s[mbase + j], rowsum[j]);
    }
    __syncthreads();
    if (tid < MTILE) {
        int t = t0 + tid;
        if (t < T_) scores[(size_t)bl * T_ + t] = score_s[tid] + attn_b[l];
    }
}

// ---------------------------------------------------------------------------
// Kernel B: per-(b,l) softmax + stable rank mask + pool-coverage coefficients.
// ---------------------------------------------------------------------------
__global__ void __launch_bounds__(256)
mask_softmax_kernel(const float* __restrict__ scores,        // [B*L, T]
                    const float* __restrict__ layer_weights, // [L]
                    float* __restrict__ coef)                // [B*L, T]
{
    __shared__ float s[T_];
    __shared__ float red[256];
    const int tid = threadIdx.x;
    const int bl  = blockIdx.x;
    const int l   = bl % L_;

    for (int t = tid; t < T_; t += 256) s[t] = scores[(size_t)bl * T_ + t];
    __syncthreads();

    float mx = -3.402823466e+38f;
    for (int t = tid; t < T_; t += 256) mx = fmaxf(mx, s[t]);
    red[tid] = mx; __syncthreads();
    for (int st = 128; st > 0; st >>= 1) {
        if (tid < st) red[tid] = fmaxf(red[tid], red[tid + st]);
        __syncthreads();
    }
    mx = red[0]; __syncthreads();

    float sum = 0.0f;
    for (int t = tid; t < T_; t += 256) sum += expf(s[t] - mx);
    red[tid] = sum; __syncthreads();
    for (int st = 128; st > 0; st >>= 1) {
        if (tid < st) red[tid] += red[tid + st];
        __syncthreads();
    }
    float inv = 1.0f / red[0];
    float lw  = layer_weights[l] * (1.0f / (float)T_);

    for (int t = tid; t < T_; t += 256) {
        float st_ = s[t];
        int cnt = 0;
        for (int j = 0; j < T_; ++j) {      // stable rank (ties by index)
            float sj = s[j];
            cnt += ((sj < st_) || (sj == st_ && j < t)) ? 1 : 0;
        }
        float c = 0.0f;
        if (cnt >= NMASK) {
            int lo = t < 2 ? t : 2;                         // pool coverage
            int hi = (T_ - 1 - t) < 2 ? (T_ - 1 - t) : 2;   // (count_include_pad)
            c = expf(st_ - mx) * inv * lw * (float)(lo + hi + 1) * 0.2f;
        }
        coef[(size_t)bl * T_ + t] = c;
    }
}

// ---------------------------------------------------------------------------
// Kernel C: memory-bound weighted reduction over t; skips masked rows
// (uniform branch -> loads skipped too, ~30% bandwidth saved).
// ---------------------------------------------------------------------------
__global__ void __launch_bounds__(256)
weighted_sum_kernel(const float* __restrict__ X,     // [B*L, T, D]
                    const float* __restrict__ coef,  // [B*L, T]
                    float* __restrict__ partial)     // [B*L, NCHUNKS, D]
{
    const int tid = threadIdx.x;
    const int bl  = blockIdx.x;
    const int ch  = blockIdx.y;
    const int d0  = tid * 2;

    float2 acc = make_float2(0.0f, 0.0f);
    const int tbeg = ch * TCHUNK;
    const int tend = tbeg + TCHUNK;
    for (int t = tbeg; t < tend; ++t) {
        float w = coef[(size_t)bl * T_ + t];
        if (w != 0.0f) {
            float2 v = *(const float2*)(X + ((size_t)bl * T_ + t) * D_ + d0);
            acc.x += w * v.x;
            acc.y += w * v.y;
        }
    }
    *(float2*)(partial + ((size_t)bl * NCHUNKS + ch) * D_ + d0) = acc;
}

// ---------------------------------------------------------------------------
// Kernel D: final deterministic reduction over L*NCHUNKS partials -> out[B,D]
// ---------------------------------------------------------------------------
__global__ void __launch_bounds__(512)
final_reduce_kernel(const float* __restrict__ partial, // [B, L*NCHUNKS, D]
                    float* __restrict__ out)           // [B, D]
{
    const int b = blockIdx.x;
    const int d = threadIdx.x;
    float sum = 0.0f;
    for (int i = 0; i < L_ * NCHUNKS; ++i)
        sum += partial[((size_t)b * L_ * NCHUNKS + i) * D_ + d];
    out[(size_t)b * D_ + d] = sum;
}

extern "C" void kernel_launch(void* const* d_in, const int* in_sizes, int n_in,
                              void* d_out, int out_size, void* d_ws, size_t ws_size,
                              hipStream_t stream) {
    const float* X  = (const float*)d_in[0];   // layer_reps [B,L,T,D]
    const float* Wl = (const float*)d_in[1];   // lin_w      [L,D,D]
    const float* lb = (const float*)d_in[2];   // lin_b      [L,D]
    const float* aw = (const float*)d_in[3];   // attn_w     [L,D]
    const float* ab = (const float*)d_in[4];   // attn_b     [L]
    const float* lw = (const float*)d_in[5];   // layer_weights [L]
    float* out = (float*)d_out;                // [B,D]

    float* scores  = (float*)d_ws;                                   // B*L*T floats
    float* coef    = scores + (size_t)B_ * L_ * T_;                   // B*L*T floats
    float* partial = coef   + (size_t)B_ * L_ * T_;                   // B*L*NCHUNKS*D floats
    unsigned short* Wt = (unsigned short*)(partial + (size_t)B_ * L_ * NCHUNKS * D_); // L*D*D bf16

    wt_transpose_kernel<<<dim3(16, 16, L_), 256, 0, stream>>>(Wl, Wt);
    dim3 gA((T_ + MTILE - 1) / MTILE, B_ * L_);
    score_gemm_kernel<<<gA, 256, 0, stream>>>(X, Wt, lb, aw, ab, scores);
    mask_softmax_kernel<<<B_ * L_, 256, 0, stream>>>(scores, lw, coef);
    weighted_sum_kernel<<<dim3(B_ * L_, NCHUNKS), 256, 0, stream>>>(X, coef, partial);
    final_reduce_kernel<<<B_, 512, 0, stream>>>(partial, out);
}
